// MoEFusionTransformer_88158498718386
// MI455X (gfx1250) — matrix operand order repaired and checked by reading, not compiled, not run on hardware
//
#include <hip/hip_runtime.h>
#include <math.h>

// ---------------------------------------------------------------------------
// Types for CDNA5 WMMA (wave32): v_wmma_f32_16x16x32_bf16
// ---------------------------------------------------------------------------
typedef __bf16 bf16;
typedef bf16  bf16x4  __attribute__((ext_vector_type(4)));
typedef bf16  bf16x8  __attribute__((ext_vector_type(8)));
typedef bf16  bf16x16 __attribute__((ext_vector_type(16)));
typedef float f32x8   __attribute__((ext_vector_type(8)));

union FragAB { bf16x16 v; bf16x8 h[2]; };

#define TILE_M 128
#define TILE_N 128
#define TILE_K 32
#define LDSS   40    // TILE_K + 8 bf16 pad; row = 80 bytes (16B aligned)

enum { EP_BIAS = 1, EP_RES = 2, EP_SCALE = 4, EP_ACC = 8 };

// ---------------------------------------------------------------------------
// C[M,N] = op( A[M,K] @ W[N,K]^T ), fp32 in/out, bf16 WMMA with f32 accum.
// Block tile 128x128, BK=32, 256 threads = 8 wave32 (4x2 wave grid), each
// wave computes a 32x64 sub-tile = 2x4 WMMA fragments (8 WMMA / k-step).
// LDS double-buffered; global loads register-staged; MODE is compile-time
// so each epilogue variant is branch-free with batched loads.
// All problem dims are multiples of the tile sizes (4096/1024/2048).
// ---------------------------------------------------------------------------
template <int MODE>
__global__ __launch_bounds__(256)
void gemm_xt_wmma(const float* __restrict__ A, const float* __restrict__ W,
                  const float* __restrict__ bias, const float* __restrict__ res,
                  const float* __restrict__ rowscale, int rs_stride,
                  float* __restrict__ C, int M, int N, int K)
{
    __shared__ __align__(16) bf16 Asb[2][TILE_M][LDSS];
    __shared__ __align__(16) bf16 Bsb[2][TILE_N][LDSS];

    const int tid  = threadIdx.x;
    const int lane = tid & 31;
    const int wave = tid >> 5;              // 0..7
    const int wm   = (wave >> 1) * 32;      // 0,32,64,96  (M within block)
    const int wn   = (wave & 1) * 64;       // 0,64        (N within block)
    const int m16  = lane & 15;
    const int kh   = lane >> 4;             // lane half -> K sub-range

    const int n0 = blockIdx.x * TILE_N;
    const int m0 = blockIdx.y * TILE_M;

    // staging coordinates: 1024 float4 per tile, 4 per thread
    const int srow0 = tid >> 3;
    const int scol  = (tid & 7) * 4;

    f32x8 acc[2][4];
#pragma unroll
    for (int mi = 0; mi < 2; ++mi)
#pragma unroll
        for (int ni = 0; ni < 4; ++ni)
#pragma unroll
            for (int r = 0; r < 8; ++r) acc[mi][ni][r] = 0.0f;

    float4 ar[4], br[4];

    // ---- prologue: stage tile 0 (all loads issued before any LDS store) ----
#pragma unroll
    for (int i = 0; i < 4; ++i) {
        const int row = srow0 + i * 32;
        ar[i] = *(const float4*)(A + (size_t)(m0 + row) * K + scol);
        br[i] = *(const float4*)(W + (size_t)(n0 + row) * K + scol);
    }
#pragma unroll
    for (int i = 0; i < 4; ++i) {
        const int row = srow0 + i * 32;
        bf16x4 ha, hb;
        ha[0] = (bf16)ar[i].x; ha[1] = (bf16)ar[i].y; ha[2] = (bf16)ar[i].z; ha[3] = (bf16)ar[i].w;
        hb[0] = (bf16)br[i].x; hb[1] = (bf16)br[i].y; hb[2] = (bf16)br[i].z; hb[3] = (bf16)br[i].w;
        *(bf16x4*)&Asb[0][row][scol] = ha;
        *(bf16x4*)&Bsb[0][row][scol] = hb;
    }

    const int nk = K / TILE_K;
    for (int kt = 0; kt < nk; ++kt) {
        const int buf = kt & 1;

        // ---- issue global loads for tile kt+1 (overlap with WMMA below) ----
        if (kt + 1 < nk) {
            const int k1 = (kt + 1) * TILE_K;
#pragma unroll
            for (int i = 0; i < 4; ++i) {
                const int row = srow0 + i * 32;
                ar[i] = *(const float4*)(A + (size_t)(m0 + row) * K + k1 + scol);
                br[i] = *(const float4*)(W + (size_t)(n0 + row) * K + k1 + scol);
            }
            if (kt + 2 < nk) {
                const int k2 = (kt + 2) * TILE_K;
                __builtin_prefetch(A + (size_t)(m0 + (tid >> 1)) * K + k2 + (tid & 1) * 16, 0, 3);
                __builtin_prefetch(W + (size_t)(n0 + (tid >> 1)) * K + k2 + (tid & 1) * 16, 0, 3);
            }
        }

        __syncthreads();   // LDS buffer `buf` fully written

        // ---- fragments (ISA 16-bit A 16x32 / B 32x16 wave32 layouts) ----
        FragAB af[2], bfr[4];
#pragma unroll
        for (int mi = 0; mi < 2; ++mi) {
            const int row = wm + mi * 16 + m16;
            // lanes 0-15: K 0..7 & 16..23 ; lanes 16-31: K 8..15 & 24..31
            af[mi].h[0] = *(const bf16x8*)&Asb[buf][row][kh * 8];
            af[mi].h[1] = *(const bf16x8*)&Asb[buf][row][16 + kh * 8];
        }
#pragma unroll
        for (int ni = 0; ni < 4; ++ni) {
            const int row = wn + ni * 16 + m16;
            // lanes 0-15: K 0..7 & 16..23 ; lanes 16-31: K 8..15 & 24..31
            bfr[ni].h[0] = *(const bf16x8*)&Bsb[buf][row][kh * 8];
            bfr[ni].h[1] = *(const bf16x8*)&Bsb[buf][row][16 + kh * 8];
        }

#pragma unroll
        for (int mi = 0; mi < 2; ++mi)
#pragma unroll
            for (int ni = 0; ni < 4; ++ni)
                acc[mi][ni] = __builtin_amdgcn_wmma_f32_16x16x32_bf16(
                    false, af[mi].v, false, bfr[ni].v,
                    (short)0, acc[mi][ni], false, false);

        // ---- convert + store staged tile kt+1 into the other buffer ----
        if (kt + 1 < nk) {
            const int nb = buf ^ 1;
#pragma unroll
            for (int i = 0; i < 4; ++i) {
                const int row = srow0 + i * 32;
                bf16x4 ha, hb;
                ha[0] = (bf16)ar[i].x; ha[1] = (bf16)ar[i].y; ha[2] = (bf16)ar[i].z; ha[3] = (bf16)ar[i].w;
                hb[0] = (bf16)br[i].x; hb[1] = (bf16)br[i].y; hb[2] = (bf16)br[i].z; hb[3] = (bf16)br[i].w;
                *(bf16x4*)&Asb[nb][row][scol] = ha;
                *(bf16x4*)&Bsb[nb][row][scol] = hb;
            }
        }
        // next iteration's __syncthreads() publishes these stores
    }

    // ---- epilogue (branch-free per MODE specialization) ----
    // C/D layout: VGPR r -> lanes0-15 M=r, lanes16-31 M=8+r; N = lane&15.
#pragma unroll
    for (int mi = 0; mi < 2; ++mi) {
#pragma unroll
        for (int ni = 0; ni < 4; ++ni) {
            const int ncol   = n0 + wn + ni * 16 + m16;
            const int mbase  = m0 + wm + mi * 16 + kh * 8;
            const float bv   = (MODE & EP_BIAS) ? bias[ncol] : 0.0f;

            float resv[8];
            if (MODE & EP_RES) {
#pragma unroll
                for (int r = 0; r < 8; ++r)
                    resv[r] = res[(size_t)(mbase + r) * N + ncol];
            }
            float cprev[8];
            if (MODE & EP_ACC) {
#pragma unroll
                for (int r = 0; r < 8; ++r)
                    cprev[r] = C[(size_t)(mbase + r) * N + ncol];
            }
            float scl[8];
            if (MODE & EP_SCALE) {
#pragma unroll
                for (int r = 0; r < 8; ++r)
                    scl[r] = rowscale[(size_t)(mbase + r) * rs_stride];
            }

#pragma unroll
            for (int r = 0; r < 8; ++r) {
                float val = acc[mi][ni][r] + bv;
                if (MODE & EP_RES)   val += resv[r];
                if (MODE & EP_SCALE) val *= scl[r];
                if (MODE & EP_ACC)   val += cprev[r];
                C[(size_t)(mbase + r) * N + ncol] = val;
            }
        }
    }
}

// ---------------------------------------------------------------------------
// Row LayerNorm: one 256-thread block per row. Handles plain LN (in2==null),
// concat LN over [in|in2] (width 2*halfD), and premultiplied input (ln(2p)).
// Safe in-place: inputs are fully staged in LDS before any write.
// ---------------------------------------------------------------------------
__global__ __launch_bounds__(256)
void ln_rows(const float* __restrict__ in, const float* __restrict__ in2,
             const float* __restrict__ g, const float* __restrict__ b,
             float* __restrict__ out, int D, int halfD, float premul)
{
    __shared__ float xbuf[2048];
    __shared__ float rs[256];
    __shared__ float rq[256];

    const int row = blockIdx.x;
    const int tid = threadIdx.x;
    const int inStride = in2 ? halfD : D;

    float s = 0.0f, sq = 0.0f;
    for (int i = tid; i < D; i += 256) {
        float v;
        if (in2 && i >= halfD) v = in2[(size_t)row * halfD + (i - halfD)];
        else                   v = in [(size_t)row * inStride + i];
        v *= premul;
        xbuf[i] = v;
        s += v; sq += v * v;
    }
    rs[tid] = s; rq[tid] = sq;
    __syncthreads();
#pragma unroll
    for (int off = 128; off > 0; off >>= 1) {
        if (tid < off) { rs[tid] += rs[tid + off]; rq[tid] += rq[tid + off]; }
        __syncthreads();
    }
    const float mean = rs[0] / (float)D;
    const float var  = rq[0] / (float)D - mean * mean;
    const float inv  = rsqrtf(var + 1e-5f);

    for (int i = tid; i < D; i += 256)
        out[(size_t)row * D + i] = (xbuf[i] - mean) * inv * g[i] + b[i];
}

// ---------------------------------------------------------------------------
// Gating: logits_e = -||[t1_b,t2_b] - gate_w_e||_2 (expanded form), top-2
// softmax scattered into gating[b, 0..7]. One block per row.
// ---------------------------------------------------------------------------
__global__ __launch_bounds__(256)
void gating_kernel(const float* __restrict__ t1, const float* __restrict__ t2,
                   const float* __restrict__ gw, float* __restrict__ gating)
{
    __shared__ float red[17][257];
    const int row = blockIdx.x;
    const int tid = threadIdx.x;

    float xx = 0.0f;
    float dots[8], gn[8];
#pragma unroll
    for (int e = 0; e < 8; ++e) { dots[e] = 0.0f; gn[e] = 0.0f; }

    for (int i = tid; i < 2048; i += 256) {
        const float x = (i < 1024) ? t1[(size_t)row * 1024 + i]
                                   : t2[(size_t)row * 1024 + (i - 1024)];
        xx += x * x;
#pragma unroll
        for (int e = 0; e < 8; ++e) {
            const float w = gw[(size_t)e * 2048 + i];
            dots[e] += x * w;
            gn[e]   += w * w;
        }
    }
    red[0][tid] = xx;
#pragma unroll
    for (int e = 0; e < 8; ++e) { red[1 + e][tid] = dots[e]; red[9 + e][tid] = gn[e]; }
    __syncthreads();
#pragma unroll
    for (int off = 128; off > 0; off >>= 1) {
        if (tid < off)
#pragma unroll
            for (int j = 0; j < 17; ++j) red[j][tid] += red[j][tid + off];
        __syncthreads();
    }
    if (tid == 0) {
        float lg[8];
#pragma unroll
        for (int e = 0; e < 8; ++e) {
            const float sqv = red[0][0] - 2.0f * red[1 + e][0] + red[9 + e][0];
            lg[e] = -sqrtf(fmaxf(sqv, 0.0f));
        }
        int i0 = 0;
        for (int e = 1; e < 8; ++e) if (lg[e] > lg[i0]) i0 = e;
        int i1 = (i0 == 0) ? 1 : 0;
        for (int e = 0; e < 8; ++e) if (e != i0 && lg[e] > lg[i1]) i1 = e;
        const float m  = lg[i0];
        const float e0 = expf(lg[i0] - m);
        const float e1 = expf(lg[i1] - m);
        const float sm = e0 + e1;
        for (int e = 0; e < 8; ++e) {
            float v = 0.0f;
            if (e == i0) v = e0 / sm;
            else if (e == i1) v = e1 / sm;
            gating[(size_t)row * 8 + e] = v;
        }
    }
}

// ---------------------------------------------------------------------------
// Host-side orchestration
// ---------------------------------------------------------------------------
static inline void launch_gemm(const float* A, const float* W, const float* bias,
                               const float* res, const float* rowscale, int rs_stride,
                               float* C, int M, int N, int K, int mode, hipStream_t s)
{
    dim3 grid(N / TILE_N, M / TILE_M);
    switch (mode) {
    case EP_BIAS:
        gemm_xt_wmma<EP_BIAS><<<grid, 256, 0, s>>>(A, W, bias, res, rowscale, rs_stride, C, M, N, K);
        break;
    case EP_BIAS | EP_RES:
        gemm_xt_wmma<EP_BIAS | EP_RES><<<grid, 256, 0, s>>>(A, W, bias, res, rowscale, rs_stride, C, M, N, K);
        break;
    case EP_BIAS | EP_SCALE:
        gemm_xt_wmma<EP_BIAS | EP_SCALE><<<grid, 256, 0, s>>>(A, W, bias, res, rowscale, rs_stride, C, M, N, K);
        break;
    default:
        gemm_xt_wmma<EP_BIAS | EP_SCALE | EP_ACC><<<grid, 256, 0, s>>>(A, W, bias, res, rowscale, rs_stride, C, M, N, K);
        break;
    }
}

extern "C" void kernel_launch(void* const* d_in, const int* in_sizes, int n_in,
                              void* d_out, int out_size, void* d_ws, size_t ws_size,
                              hipStream_t stream)
{
    (void)in_sizes; (void)n_in; (void)out_size; (void)ws_size;

    const int B = 4096, In = 1024, D = 1024, E = 8;

    const float* t1      = (const float*)d_in[0];
    const float* t2      = (const float*)d_in[1];
    const float* proj_w  = (const float*)d_in[2];
    const float* proj_b  = (const float*)d_in[3];
    const float* ln1_g   = (const float*)d_in[4];
    const float* ln1_b   = (const float*)d_in[5];
    const float* ca_in_w = (const float*)d_in[6];
    const float* ca_in_b = (const float*)d_in[7];
    const float* ca_o_w  = (const float*)d_in[8];
    const float* ca_o_b  = (const float*)d_in[9];
    const float* ln2_g   = (const float*)d_in[10];
    const float* ln2_b   = (const float*)d_in[11];
    const float* sa_in_w = (const float*)d_in[12];
    const float* sa_in_b = (const float*)d_in[13];
    const float* sa_o_w  = (const float*)d_in[14];
    const float* sa_o_b  = (const float*)d_in[15];
    const float* ln3_g   = (const float*)d_in[16];
    const float* ln3_b   = (const float*)d_in[17];
    const float* ffn_w   = (const float*)d_in[18];
    const float* ffn_b   = (const float*)d_in[19];
    const float* gate_w  = (const float*)d_in[20];
    const float* out_w   = (const float*)d_in[21];
    const float* out_b   = (const float*)d_in[22];
    const float* lno_g   = (const float*)d_in[23];
    const float* lno_b   = (const float*)d_in[24];
    float* outp = (float*)d_out;

    // workspace carve-up (fp32)
    float* ws = (float*)d_ws;
    size_t o = 0;
    float* gating = ws + o; o += (size_t)B * E;
    float* x1     = ws + o; o += (size_t)B * D;   // x1 (post-LN1)
    float* x2     = ws + o; o += (size_t)B * D;   // x2 -> later s
    float* tA     = ws + o; o += (size_t)B * D;   // v, then v2
    float* tB     = ws + o; o += (size_t)B * D;   // a_pre -> a
    float* fused  = ws + o; o += (size_t)B * 2 * D;
    float* acc    = ws + o; o += (size_t)B * D;   // gated expert accumulator
    float* p      = ws + o; o += (size_t)B * D;   // output projection

    // ---- gating ----
    gating_kernel<<<B, 256, 0, stream>>>(t1, t2, gate_w, gating);

    // ---- dense expert loop ----
    for (int e = 0; e < E; ++e) {
        const float* pw   = proj_w  + (size_t)e * D * In;
        const float* pb   = proj_b  + (size_t)e * D;
        const float* g1   = ln1_g   + (size_t)e * D;
        const float* b1   = ln1_b   + (size_t)e * D;
        const float* cavw = ca_in_w + (size_t)e * 3 * D * D + (size_t)2 * D * D; // V slice
        const float* cavb = ca_in_b + (size_t)e * 3 * D + 2 * D;
        const float* cow  = ca_o_w  + (size_t)e * D * D;
        const float* cob  = ca_o_b  + (size_t)e * D;
        const float* g2   = ln2_g   + (size_t)e * D;
        const float* b2   = ln2_b   + (size_t)e * D;
        const float* savw = sa_in_w + (size_t)e * 3 * D * D + (size_t)2 * D * D; // V slice
        const float* savb = sa_in_b + (size_t)e * 3 * D + 2 * D;
        const float* sow  = sa_o_w  + (size_t)e * D * D;
        const float* sob  = sa_o_b  + (size_t)e * D;
        const float* g3   = ln3_g   + (size_t)e * 2 * D;
        const float* b3   = ln3_b   + (size_t)e * 2 * D;
        const float* fw   = ffn_w   + (size_t)e * D * 2 * D;
        const float* fb   = ffn_b   + (size_t)e * D;

        // x1 = LN1(t1 @ pw^T + pb)
        launch_gemm(t1, pw, pb, nullptr, nullptr, 0, x1, B, D, In, EP_BIAS, stream);
        ln_rows<<<B, 256, 0, stream>>>(x1, nullptr, g1, b1, x1, D, 0, 1.0f);
        // x2 = LN1(t2 @ pw^T + pb)
        launch_gemm(t2, pw, pb, nullptr, nullptr, 0, x2, B, D, In, EP_BIAS, stream);
        ln_rows<<<B, 256, 0, stream>>>(x2, nullptr, g1, b1, x2, D, 0, 1.0f);
        // v = x2 @ cavw^T + cavb
        launch_gemm(x2, cavw, cavb, nullptr, nullptr, 0, tA, B, D, D, EP_BIAS, stream);
        // a = LN2(x1 + v @ cow^T + cob)
        launch_gemm(tA, cow, cob, x1, nullptr, 0, tB, B, D, D, EP_BIAS | EP_RES, stream);
        ln_rows<<<B, 256, 0, stream>>>(tB, nullptr, g2, b2, tB, D, 0, 1.0f);
        // v2 = a @ savw^T + savb
        launch_gemm(tB, savw, savb, nullptr, nullptr, 0, tA, B, D, D, EP_BIAS, stream);
        // s = LN2(a + v2 @ sow^T + sob)   (x2 is dead; reuse as s)
        launch_gemm(tA, sow, sob, tB, nullptr, 0, x2, B, D, D, EP_BIAS | EP_RES, stream);
        ln_rows<<<B, 256, 0, stream>>>(x2, nullptr, g2, b2, x2, D, 0, 1.0f);
        // fused = LN3(concat(x1, s))
        ln_rows<<<B, 256, 0, stream>>>(x1, x2, g3, b3, fused, 2 * D, D, 1.0f);
        // acc (+)= gating[:,e] * (fused @ fw^T + fb)
        const int mode = EP_BIAS | EP_SCALE | (e ? EP_ACC : 0);
        launch_gemm(fused, fw, fb, nullptr, gating + e, E, acc, B, D, 2 * D, mode, stream);
    }

    // ---- output projection + final LN(2p) ----
    launch_gemm(acc, out_w, out_b, nullptr, nullptr, 0, p, B, D, D, EP_BIAS, stream);
    ln_rows<<<B, 256, 0, stream>>>(p, nullptr, lno_g, lno_b, outp, D, 0, 2.0f);
}
